// MultiHeadAttention_68736656605652
// MI455X (gfx1250) — matrix operand order repaired and checked by reading, not compiled, not run on hardware
//
#include <hip/hip_runtime.h>

#define D_MODEL 1024
#define N_HEADS 16
#define D_HEAD  64
#define B_SZ    2
#define T_SEQ   2048
#define M_TOK   (B_SZ * T_SEQ)   // 4096

typedef __attribute__((ext_vector_type(16))) _Float16 v16h;
typedef __attribute__((ext_vector_type(8)))  float    v8f;

union U16h { v16h v; uint4 q[2]; unsigned u[8]; _Float16 h[16]; };
union U8f  { v8f v; float f[8]; };

static __device__ __forceinline__ unsigned pack2(float a, float b) {
    union { _Float16 h[2]; unsigned u; } t;
    t.h[0] = (_Float16)a; t.h[1] = (_Float16)b;
    return t.u;
}

static __device__ __forceinline__ v8f wmma16(v16h a, v16h b, v8f c) {
    return __builtin_amdgcn_wmma_f32_16x16x32_f16(
        false, a, false, b, (short)0, c, false, false);
}

// Half-wave swap (lane <-> lane^16) in one VALU op: v_permlanex16_b32 with
// identity lane selects. Replaces ds_bpermute-based __shfl_xor lowering.
static __device__ __forceinline__ unsigned xor16_u32(unsigned x) {
    return (unsigned)__builtin_amdgcn_permlanex16(
        (int)x, (int)x, 0x76543210, (int)0xfedcba98, false, false);
}
static __device__ __forceinline__ float xor16_f32(float x) {
    return __uint_as_float(xor16_u32(__float_as_uint(x)));
}

// Async global -> LDS copy, 16 bytes per lane, tracked by ASYNCcnt.
static __device__ __forceinline__ void async_b128(unsigned lds_off, const _Float16* g) {
    unsigned long long ga = (unsigned long long)(uintptr_t)g;
    asm volatile("global_load_async_to_lds_b128 %0, %1, off"
                 :: "v"(lds_off), "v"(ga) : "memory");
}
#define WAIT_ASYNC(n) asm volatile("s_wait_asynccnt " #n ::: "memory")

static __device__ __forceinline__ unsigned lds_off(const void* p) {
    return (unsigned)(uintptr_t)p;   // low 32 bits of flat LDS address = LDS offset
}

// ---------------------------------------------------------------------------
// 1) f32 -> f16 conversion
// ---------------------------------------------------------------------------
__global__ void cvt_f32_f16(const float* __restrict__ in, _Float16* __restrict__ out, int n) {
    int i = blockIdx.x * blockDim.x + threadIdx.x;
    int stride = gridDim.x * blockDim.x;
    for (; i < n; i += stride) out[i] = (_Float16)in[i];
}

// ---------------------------------------------------------------------------
// 2/5) GEMM: Y[tok, ch] = sum_k X[tok,k] * W[ch,k]   (y = x @ W^T)
//      Block = 4 waves -> 128ch x 128tok tile. W/X k-slices staged to LDS via
//      async copies, double buffered; fragments read back with ds_load_b128.
// ---------------------------------------------------------------------------
template<bool F16OUT>
__global__ __launch_bounds__(128) void gemm_xwT(
    const _Float16* __restrict__ W,   // [1024 out, 1024 in]
    const _Float16* __restrict__ X,   // [M_TOK, 1024]
    void* __restrict__ Yout)          // [M_TOK, 1024] f16 or f32
{
    __shared__ __align__(16) _Float16 shW[2][128 * 32];  // 8 KiB per buffer
    __shared__ __align__(16) _Float16 shX[2][128 * 32];

    const int tid  = threadIdx.x;
    const int lane = tid & 31;
    const int wave = tid >> 5;
    const int l15  = lane & 15;
    const bool lo  = lane < 16;
    const int K = D_MODEL;
    const int chBlock  = blockIdx.x * 128;
    const int tokBlock = blockIdx.y * 128;
    const int chW  = (wave & 1) * 64;
    const int tokW = (wave >> 1) * 64;

    const unsigned lW[2] = { lds_off(&shW[0][0]), lds_off(&shW[1][0]) };
    const unsigned lX[2] = { lds_off(&shX[0][0]), lds_off(&shX[1][0]) };

    auto stage = [&](int s, int k0) {
        const _Float16* gw = W + (size_t)(chBlock  + tid) * K + k0;
        const _Float16* gx = X + (size_t)(tokBlock + tid) * K + k0;
        const unsigned lw = lW[s] + tid * 64;   // 32 halfs = 64 B per row
        const unsigned lx = lX[s] + tid * 64;
#pragma unroll
        for (int j = 0; j < 4; j++) {
            async_b128(lw + j * 16, gw + j * 8);
            async_b128(lx + j * 16, gx + j * 8);
        }
    };

    v8f vz = {};
    v8f acc[4][4];
#pragma unroll
    for (int i = 0; i < 4; i++)
#pragma unroll
        for (int j = 0; j < 4; j++) acc[i][j] = vz;

    stage(0, 0);
    for (int k0 = 0; k0 < K; k0 += 32) {
        const int s = (k0 >> 5) & 1;
        __syncthreads();                      // prev compute done: safe to overwrite s^1
        if (k0 + 32 < K) {
            stage(s ^ 1, k0 + 32);
            WAIT_ASYNC(8);                    // 8 newest in flight -> buffer s complete
        } else {
            WAIT_ASYNC(0);
        }
        __syncthreads();                      // all waves' staging for s visible

        U16h a[4], b[4];
#pragma unroll
        for (int ct = 0; ct < 4; ct++) {
            const _Float16* p = &shW[s][(chW + ct * 16 + l15) * 32 + (lo ? 0 : 8)];
            a[ct].q[0] = *(const uint4*)p;
            a[ct].q[1] = *(const uint4*)(p + 16);
        }
#pragma unroll
        for (int tt = 0; tt < 4; tt++) {
            const _Float16* p = &shX[s][(tokW + tt * 16 + l15) * 32 + (lo ? 0 : 8)];
            b[tt].q[0] = *(const uint4*)p;
            b[tt].q[1] = *(const uint4*)(p + 16);
        }
#pragma unroll
        for (int ct = 0; ct < 4; ct++)
#pragma unroll
            for (int tt = 0; tt < 4; tt++)
                acc[ct][tt] = wmma16(a[ct].v, b[tt].v, acc[ct][tt]);
    }

#pragma unroll
    for (int ct = 0; ct < 4; ct++) {
#pragma unroll
        for (int tt = 0; tt < 4; tt++) {
            const int tok = tokBlock + tokW + tt * 16 + l15;
            const int ch0 = chBlock + chW + ct * 16 + (lo ? 0 : 8);
            U8f c; c.v = acc[ct][tt];
            if constexpr (F16OUT) {
                unsigned pk[4];
#pragma unroll
                for (int j = 0; j < 4; j++) pk[j] = pack2(c.f[2 * j], c.f[2 * j + 1]);
                *(uint4*)((_Float16*)Yout + (size_t)tok * D_MODEL + ch0) =
                    make_uint4(pk[0], pk[1], pk[2], pk[3]);
            } else {
                float* o = (float*)Yout + (size_t)tok * D_MODEL + ch0;
                *(float4*)(o)     = make_float4(c.f[0], c.f[1], c.f[2], c.f[3]);
                *(float4*)(o + 4) = make_float4(c.f[4], c.f[5], c.f[6], c.f[7]);
            }
        }
    }
}

// ---------------------------------------------------------------------------
// 3) V transpose: Vtok[b*T + t, h*64 + d] -> Vt[(b*H+h)*64 + d, t]
// ---------------------------------------------------------------------------
__global__ __launch_bounds__(256) void transpose_v(
    const _Float16* __restrict__ Vtok, _Float16* __restrict__ Vt)
{
    __shared__ _Float16 tile[16][72];
    const int tT = blockIdx.x, dT = blockIdx.y, bh = blockIdx.z;
    const int b = bh >> 4, h = bh & 15;
    const int t0 = tT * 64, d0 = dT * 16;
    const int tid = threadIdx.x;

    const int dd = tid & 15, tt0 = tid >> 4;
#pragma unroll
    for (int i = 0; i < 4; i++) {
        int t = tt0 + i * 16;
        tile[dd][t] = Vtok[(size_t)(b * T_SEQ + t0 + t) * D_MODEL + h * D_HEAD + d0 + dd];
    }
    __syncthreads();
    const int t2 = tid & 63, d2 = tid >> 6;
#pragma unroll
    for (int i = 0; i < 4; i++) {
        int d = d2 + i * 4;
        Vt[((size_t)bh * D_HEAD + d0 + d) * T_SEQ + t0 + t2] = tile[d][t2];
    }
}

// ---------------------------------------------------------------------------
// 4) Causal flash attention. 4 waves/block share one (b,h); K/V tiles staged
//    once per block via async copies, double buffered. Per wave:
//    S^T = K * Q^T (lane owns a query -> per-lane online softmax, exp2 domain),
//    O^T = V^T * P (P built with one v_permlanex16 half-wave swap).
// ---------------------------------------------------------------------------
__global__ __launch_bounds__(128) void attn_kernel(
    const _Float16* __restrict__ Qtok,  // [M_TOK, 1024]
    const _Float16* __restrict__ Ktok,  // [M_TOK, 1024]
    const _Float16* __restrict__ Vtg,   // [B*H, 64, T]
    _Float16* __restrict__ Attn)        // [M_TOK, 1024]
{
    __shared__ __align__(16) _Float16 shK[2][32 * 64];  // 4 KiB per buffer
    __shared__ __align__(16) _Float16 shV[2][64 * 32];  // 4 KiB per buffer

    const int tid  = threadIdx.x;
    const int lane = tid & 31;
    const int wave = tid >> 5;
    const int gid  = blockIdx.x * 4 + wave;   // 0..4095
    const int qt = gid & 127;
    const int bh = gid >> 7;
    const int b = bh >> 4, h = bh & 15;
    const int qBase = qt * 16;
    const int l15 = lane & 15;
    const bool lo = lane < 16;
    const int qrow = qBase + l15;

    const int qtHiBase = ((blockIdx.x * 4) & 127) * 16 + 48;  // highest wave's qBase
    const int nktMax = (qtHiBase + 47) >> 5;                  // block-uniform step count
    const int nkt    = (qBase + 47) >> 5;                     // this wave's step count
    const int nFull  = qBase >> 5;                            // steps needing no mask

    const size_t baseK = (size_t)(b * T_SEQ) * D_MODEL + h * D_HEAD;
    const size_t baseV = (size_t)bh * D_HEAD * T_SEQ;

    const unsigned lK[2] = { lds_off(&shK[0][0]), lds_off(&shK[1][0]) };
    const unsigned lV[2] = { lds_off(&shV[0][0]), lds_off(&shV[1][0]) };

    // K tile: 32 keys x 64 d (row = 128 B); V tile: 64 d x 32 keys (row = 64 B)
    auto stage = [&](int s, int kb) {
        const int krow = tid >> 2, kj = tid & 3;              // 4 x 32 B per key row
        const _Float16* gk = Ktok + baseK + (size_t)(kb + krow) * D_MODEL + kj * 16;
        const unsigned lk = lK[s] + (krow * 64 + kj * 16) * 2;
        async_b128(lk,      gk);
        async_b128(lk + 16, gk + 8);
        const int drow = tid >> 1, dj = tid & 1;              // 2 x 32 B per d row
        const _Float16* gv = Vtg + baseV + (size_t)drow * T_SEQ + kb + dj * 16;
        const unsigned lv = lV[s] + (drow * 32 + dj * 16) * 2;
        async_b128(lv,      gv);
        async_b128(lv + 16, gv + 8);
    };

    // Q^T B-fragments (K-dim = d, N = query).
    // Scale folds 1/sqrt(64) AND log2(e): softmax runs in exp2 domain.
    U16h qf[2];
    const _Float16 qscale = (_Float16)(0.125f * 1.4426950408889634f);
    const size_t rowQ = (size_t)(b * T_SEQ + qrow) * D_MODEL + h * D_HEAD;
#pragma unroll
    for (int dc = 0; dc < 2; dc++) {
        const _Float16* p = Qtok + rowQ + dc * 32 + (lo ? 0 : 8);
        qf[dc].q[0] = *(const uint4*)p;
        qf[dc].q[1] = *(const uint4*)(p + 16);
#pragma unroll
        for (int e = 0; e < 16; e++) qf[dc].h[e] = qf[dc].h[e] * qscale;
    }

    v8f vz = {};
    v8f oacc[4] = {vz, vz, vz, vz};
    float m  = -__builtin_inff();
    float pl = 0.0f;

    stage(0, 0);
    for (int it = 0; it < nktMax; ++it) {
        const int kb = it * 32;
        const int s = it & 1;
        __syncthreads();                      // prev compute done: safe to overwrite s^1
        if (it + 1 < nktMax) {
            stage(s ^ 1, kb + 32);
            WAIT_ASYNC(4);
        } else {
            WAIT_ASYNC(0);
        }
        __syncthreads();

        if (it < nkt) {
            // K A-fragments from LDS (M=key, K=d)
            U16h kf[2][2];
#pragma unroll
            for (int kt = 0; kt < 2; kt++) {
                const int krow = kt * 16 + l15;
#pragma unroll
                for (int dc = 0; dc < 2; dc++) {
                    const _Float16* p = &shK[s][krow * 64 + dc * 32 + (lo ? 0 : 8)];
                    kf[kt][dc].q[0] = *(const uint4*)p;
                    kf[kt][dc].q[1] = *(const uint4*)(p + 16);
                }
            }
            v8f st0 = vz, st1 = vz;
            st0 = wmma16(kf[0][0].v, qf[0].v, st0);
            st0 = wmma16(kf[0][1].v, qf[1].v, st0);
            st1 = wmma16(kf[1][0].v, qf[0].v, st1);
            st1 = wmma16(kf[1][1].v, qf[1].v, st1);

            U8f sc[2]; sc[0].v = st0; sc[1].v = st1;
            if (it >= nFull) {                // only last 1-2 tiles need the causal mask
#pragma unroll
                for (int kt = 0; kt < 2; kt++)
#pragma unroll
                    for (int r = 0; r < 8; r++) {
                        const int key = kb + kt * 16 + r + (lo ? 0 : 8);
                        if (key > qrow) sc[kt].f[r] = -__builtin_inff();
                    }
            }
            float mt = -__builtin_inff();
#pragma unroll
            for (int kt = 0; kt < 2; kt++)
#pragma unroll
                for (int r = 0; r < 8; r++) mt = fmaxf(mt, sc[kt].f[r]);

            const float mo   = xor16_f32(mt);
            const float mnew = fmaxf(fmaxf(mt, mo), m);
            const float alpha = exp2f(m - mnew);     // exp2 domain
            m = mnew;
            pl *= alpha;
#pragma unroll
            for (int dt = 0; dt < 4; dt++) {
                U8f o; o.v = oacc[dt];
#pragma unroll
                for (int e = 0; e < 8; e++) o.f[e] *= alpha;
                oacc[dt] = o.v;
            }

            unsigned pk[2][4];
#pragma unroll
            for (int kt = 0; kt < 2; kt++) {
                float pv[8];
#pragma unroll
                for (int r = 0; r < 8; r++) {
                    pv[r] = exp2f(sc[kt].f[r] - m);  // exp2 domain
                    pl += pv[r];
                }
#pragma unroll
                for (int j = 0; j < 4; j++) pk[kt][j] = pack2(pv[2 * j], pv[2 * j + 1]);
            }
            unsigned c0[4], c1[4];
#pragma unroll
            for (int j = 0; j < 4; j++) {
                c0[j] = xor16_u32(pk[0][j]);
                c1[j] = xor16_u32(pk[1][j]);
            }
            U16h pb;
#pragma unroll
            for (int j = 0; j < 4; j++) {
                pb.u[j]     = lo ? pk[0][j] : c1[j];   // keys kb+0..7  | kb+16..23
                pb.u[4 + j] = lo ? c0[j]    : pk[1][j]; // keys kb+8..15 | kb+24..31
            }

            // V^T A-fragments from LDS (M=d, K=keys) and O^T accumulation
#pragma unroll
            for (int dt = 0; dt < 4; dt++) {
                const _Float16* p = &shV[s][(dt * 16 + l15) * 32 + (lo ? 0 : 16)];
                U16h vf;
                vf.q[0] = *(const uint4*)p;
                vf.q[1] = *(const uint4*)(p + 8);
                oacc[dt] = wmma16(vf.v, pb.v, oacc[dt]);
            }
        }
    }

    const float l = pl + xor16_f32(pl);
    const float rinv = 1.0f / l;
#pragma unroll
    for (int dt = 0; dt < 4; dt++) {
        U8f o; o.v = oacc[dt];
        unsigned pkk[4];
#pragma unroll
        for (int j = 0; j < 4; j++)
            pkk[j] = pack2(o.f[2 * j] * rinv, o.f[2 * j + 1] * rinv);
        const int d0 = dt * 16 + (lo ? 0 : 8);
        _Float16* dst = Attn + (size_t)(b * T_SEQ + qrow) * D_MODEL + h * D_HEAD + d0;
        *(uint4*)dst = make_uint4(pkk[0], pkk[1], pkk[2], pkk[3]);
    }
}

// ---------------------------------------------------------------------------
// launcher
// ---------------------------------------------------------------------------
extern "C" void kernel_launch(void* const* d_in, const int* in_sizes, int n_in,
                              void* d_out, int out_size, void* d_ws, size_t ws_size,
                              hipStream_t stream) {
    (void)in_sizes; (void)n_in; (void)out_size; (void)ws_size;
    const float* q  = (const float*)d_in[0];
    const float* k  = (const float*)d_in[1];
    const float* v  = (const float*)d_in[2];
    const float* wq = (const float*)d_in[3];
    const float* wk = (const float*)d_in[4];
    const float* wv = (const float*)d_in[5];
    const float* wo = (const float*)d_in[6];

    char* ws = (char*)d_ws;
    size_t off = 0;
    auto carve = [&](size_t bytes) {
        void* p = ws + off;
        off += (bytes + 255) & ~(size_t)255;
        return p;
    };
    const size_t szX = (size_t)M_TOK * D_MODEL * sizeof(_Float16);
    const size_t szW = (size_t)D_MODEL * D_MODEL * sizeof(_Float16);
    _Float16* qf16 = (_Float16*)carve(szX);
    _Float16* kf16 = (_Float16*)carve(szX);
    _Float16* vf16 = (_Float16*)carve(szX);
    _Float16* wq16 = (_Float16*)carve(szW);
    _Float16* wk16 = (_Float16*)carve(szW);
    _Float16* wv16 = (_Float16*)carve(szW);
    _Float16* wo16 = (_Float16*)carve(szW);
    _Float16* Qtok = (_Float16*)carve(szX);
    _Float16* Ktok = (_Float16*)carve(szX);
    _Float16* Vtok = (_Float16*)carve(szX);
    _Float16* Vt   = (_Float16*)carve(szX);
    _Float16* Attn = (_Float16*)carve(szX);

    const int nX = M_TOK * D_MODEL;
    const int nW = D_MODEL * D_MODEL;
    cvt_f32_f16<<<1024, 256, 0, stream>>>(q,  qf16, nX);
    cvt_f32_f16<<<1024, 256, 0, stream>>>(k,  kf16, nX);
    cvt_f32_f16<<<1024, 256, 0, stream>>>(v,  vf16, nX);
    cvt_f32_f16<<<512,  256, 0, stream>>>(wq, wq16, nW);
    cvt_f32_f16<<<512,  256, 0, stream>>>(wk, wk16, nW);
    cvt_f32_f16<<<512,  256, 0, stream>>>(wv, wv16, nW);
    cvt_f32_f16<<<512,  256, 0, stream>>>(wo, wo16, nW);

    dim3 gproj(D_MODEL / 128, M_TOK / 128);  // (8, 32)
    gemm_xwT<true><<<gproj, 128, 0, stream>>>(wq16, qf16, Qtok);
    gemm_xwT<true><<<gproj, 128, 0, stream>>>(wk16, kf16, Ktok);
    gemm_xwT<true><<<gproj, 128, 0, stream>>>(wv16, vf16, Vtok);

    dim3 gtr(T_SEQ / 64, D_HEAD / 16, B_SZ * N_HEADS);  // (32, 4, 32)
    transpose_v<<<gtr, 256, 0, stream>>>(Vtok, Vt);

    attn_kernel<<<(B_SZ * N_HEADS * (T_SEQ / 16)) / 4, 128, 0, stream>>>(Qtok, Ktok, Vt, Attn);

    gemm_xwT<false><<<gproj, 128, 0, stream>>>(wo16, Attn, (float*)d_out);
}
